// MLP_88699664597584
// MI455X (gfx1250) — matrix operand order repaired and checked by reading, not compiled, not run on hardware
//
#include <hip/hip_runtime.h>
#include <hip/hip_bf16.h>
#include <math.h>

// Problem dims
#define Bn   128
#define Nn   128
#define Hn   512
#define Tn   129              // N+1 scan steps
#define ROWS (Bn * Tn)        // 16512 MLP rows total
#define TM   32               // rows per workgroup (two 16-row WMMA tiles)
#define NP   144              // padded logits width (9 tiles of 16)
#define LSTR 148              // LDS logits row stride (floats)

typedef __attribute__((ext_vector_type(16))) __bf16 v16bf;
typedef __attribute__((ext_vector_type(8)))  float  v8f;

union FragU { uint4 q[2]; v16bf v; };

__device__ __forceinline__ unsigned short f2bf(float f) {
  union { float f; unsigned u; } x; x.f = f;
  unsigned r = (x.u + 0x7FFFu + ((x.u >> 16) & 1u)) >> 16;  // RNE
  return (unsigned short)r;
}

// A-matrix (16x32 bf16) fragment from row-major [16 x stride] storage.
// Lane l holds row m = l&15; K-half base = (l>>4)*8; elems 0..7 -> kbase..+7,
// elems 8..15 -> kbase+16..+23.  Two 16B loads.
__device__ __forceinline__ v16bf load_a(const unsigned short* base, int stride,
                                        int lane, int k0) {
  const unsigned short* p = base + (lane & 15) * stride + k0 + ((lane >> 4) << 3);
  FragU f;
  f.q[0] = *(const uint4*)(p);
  f.q[1] = *(const uint4*)(p + 16);
  return f.v;
}

// B-matrix (32x16 bf16) fragment.  B[k][n] = W[nrow + n][k] for h = x @ W^T with
// W row-major (out,in).  Lane l holds column n = l&15; same K layout as load_a.
__device__ __forceinline__ v16bf load_b(const unsigned short* base, int stride,
                                        int nrow, int lane, int k0) {
  const unsigned short* p = base + (nrow + (lane & 15)) * stride + k0 + ((lane >> 4) << 3);
  FragU f;
  f.q[0] = *(const uint4*)(p);
  f.q[1] = *(const uint4*)(p + 16);
  return f.v;
}

// Branch-free ELU via the native TRANS exp (co-executes with WMMA).
__device__ __forceinline__ float elu(float v) {
  return v > 0.0f ? v : (__expf(v) - 1.0f);
}

// ---------------------------------------------------------------------------
// Kernel 1: convert weights to bf16; W3 zero-padded from 129 to 144 rows.
__global__ void prep_weights(const float* __restrict__ W1, const float* __restrict__ W2,
                             const float* __restrict__ W3,
                             unsigned short* __restrict__ W1b,
                             unsigned short* __restrict__ W2b,
                             unsigned short* __restrict__ W3b) {
  const int total = Hn * Nn + Hn * Hn + NP * Hn;
  for (int i = blockIdx.x * blockDim.x + threadIdx.x; i < total;
       i += gridDim.x * blockDim.x) {
    if (i < Hn * Nn) {
      W1b[i] = f2bf(W1[i]);
    } else if (i < Hn * Nn + Hn * Hn) {
      int j = i - Hn * Nn;
      W2b[j] = f2bf(W2[j]);
    } else {
      int j = i - Hn * Nn - Hn * Hn;
      int r = j / Hn;
      W3b[j] = (r < Nn + 1) ? f2bf(W3[j]) : (unsigned short)0;
    }
  }
}

// ---------------------------------------------------------------------------
// Kernel 2: per-batch trajectory metadata.  samples = argmax(one_hot,-1);
// rank[n] = #ones before n; pos[t] = position of (t+1)-th one (128 if none);
// K = total ones.  The teacher-forced trajectory is fully determined by these.
__global__ void prep_traj(const float* __restrict__ one_hot,
                          int* __restrict__ samples, int* __restrict__ rank,
                          int* __restrict__ pos, int* __restrict__ Karr) {
  int b = blockIdx.x;
  int n = threadIdx.x;  // 128 threads
  int s = (one_hot[(b * Nn + n) * 2 + 1] > one_hot[(b * Nn + n) * 2 + 0]) ? 1 : 0;
  samples[b * Nn + n] = s;
  __syncthreads();
  if (n == 0) {
    int r = 0;
    for (int i = 0; i < Nn; ++i) {
      rank[b * Nn + i] = r;
      if (samples[b * Nn + i]) { pos[b * Nn + r] = i; ++r; }
    }
    Karr[b] = r;
    for (int t = r; t < Nn; ++t) pos[b * Nn + t] = Nn;
  }
}

// ---------------------------------------------------------------------------
// Kernel 3: fused 3-layer MLP + masked log-softmax for a 32-row tile.
// 128 threads = 4 wave32.  Each B fragment feeds two A fragments (rows 0-15
// and 16-31), halving weight traffic.  EXEC stays all-ones around all WMMAs.
__global__ __launch_bounds__(128) void mlp_fused(
    const unsigned short* __restrict__ W1b, const unsigned short* __restrict__ W2b,
    const unsigned short* __restrict__ W3b,
    const float* __restrict__ b1, const float* __restrict__ b2,
    const float* __restrict__ b3,
    const int* __restrict__ samples, const int* __restrict__ rank,
    const int* __restrict__ pos, const int* __restrict__ Karr,
    float* __restrict__ lp_rows) {
  __shared__ __align__(16) unsigned short ldsx [TM * Nn];   //  8 KB
  __shared__ __align__(16) unsigned short ldsh1[TM * Hn];   // 32 KB
  __shared__ __align__(16) unsigned short ldsh2[TM * Hn];   // 32 KB
  __shared__ __align__(16) float          ldsl [TM * LSTR]; // ~18.5 KB

  const int tid  = threadIdx.x;
  const int lane = tid & 31;
  const int wave = tid >> 5;
  const int row0 = blockIdx.x * TM;

  const int mrow = (lane >> 4) << 3;  // D-frag: row m = r + mrow (+16 for hi tile)
  const int ncol = lane & 15;         // D-frag: column within tile

  // ---- build x tile: current state at (b,t) = samples with rank < t
  for (int i = tid; i < TM * Nn; i += 128) {
    int m = i >> 7, n = i & 127;
    int grow = row0 + m;
    int b = grow / Tn, t = grow % Tn;
    int s  = samples[b * Nn + n];
    int rk = rank[b * Nn + n];
    ldsx[m * Nn + n] = f2bf((s && (rk < t)) ? 1.0f : 0.0f);
  }
  __syncthreads();

  // ---- layer 1: (32x128) @ W1^T -> 32x512, ELU.  Wave w owns cols [w*128, +128)
  for (int tp = 0; tp < 8; ++tp) {
    int c0 = wave * 128 + tp * 16;
    float bv = b1[c0 + ncol];
    v8f acc0, acc1;
    for (int r = 0; r < 8; ++r) { acc0[r] = bv; acc1[r] = bv; }
    for (int k0 = 0; k0 < Nn; k0 += 32) {
      v16bf aLo = load_a(ldsx,           Nn, lane, k0);
      v16bf aHi = load_a(ldsx + 16 * Nn, Nn, lane, k0);
      v16bf bf  = load_b(W1b, Nn, c0, lane, k0);
      acc0 = __builtin_amdgcn_wmma_f32_16x16x32_bf16(false, aLo, false, bf,
                                                     (short)0, acc0, false, false);
      acc1 = __builtin_amdgcn_wmma_f32_16x16x32_bf16(false, aHi, false, bf,
                                                     (short)0, acc1, false, false);
    }
    for (int r = 0; r < 8; ++r) {
      ldsh1[(r + mrow) * Hn + c0 + ncol]        = f2bf(elu(acc0[r]));
      ldsh1[(r + mrow + 16) * Hn + c0 + ncol]   = f2bf(elu(acc1[r]));
    }
  }
  __syncthreads();

  // ---- layer 2: (32x512) @ W2^T -> 32x512, ELU
  for (int tp = 0; tp < 8; ++tp) {
    int c0 = wave * 128 + tp * 16;
    float bv = b2[c0 + ncol];
    v8f acc0, acc1;
    for (int r = 0; r < 8; ++r) { acc0[r] = bv; acc1[r] = bv; }
    for (int k0 = 0; k0 < Hn; k0 += 32) {
      v16bf aLo = load_a(ldsh1,           Hn, lane, k0);
      v16bf aHi = load_a(ldsh1 + 16 * Hn, Hn, lane, k0);
      v16bf bf  = load_b(W2b, Hn, c0, lane, k0);
      acc0 = __builtin_amdgcn_wmma_f32_16x16x32_bf16(false, aLo, false, bf,
                                                     (short)0, acc0, false, false);
      acc1 = __builtin_amdgcn_wmma_f32_16x16x32_bf16(false, aHi, false, bf,
                                                     (short)0, acc1, false, false);
    }
    for (int r = 0; r < 8; ++r) {
      ldsh2[(r + mrow) * Hn + c0 + ncol]        = f2bf(elu(acc0[r]));
      ldsh2[(r + mrow + 16) * Hn + c0 + ncol]   = f2bf(elu(acc1[r]));
    }
  }
  __syncthreads();

  // ---- layer 3: (32x512) @ W3^T -> 32x144 logits (cols >= 129 have zero weights)
  for (int tj = wave; tj < 9; tj += 4) {
    int c0  = tj * 16;
    int col = c0 + ncol;
    float bv = (col < Nn + 1) ? b3[col] : 0.0f;
    v8f acc0, acc1;
    for (int r = 0; r < 8; ++r) { acc0[r] = bv; acc1[r] = bv; }
    for (int k0 = 0; k0 < Hn; k0 += 32) {
      v16bf aLo = load_a(ldsh2,           Hn, lane, k0);
      v16bf aHi = load_a(ldsh2 + 16 * Hn, Hn, lane, k0);
      v16bf bf  = load_b(W3b, Hn, c0, lane, k0);
      acc0 = __builtin_amdgcn_wmma_f32_16x16x32_bf16(false, aLo, false, bf,
                                                     (short)0, acc0, false, false);
      acc1 = __builtin_amdgcn_wmma_f32_16x16x32_bf16(false, aHi, false, bf,
                                                     (short)0, acc1, false, false);
    }
    for (int r = 0; r < 8; ++r) {
      ldsl[(r + mrow) * LSTR + col]      = acc0[r];
      ldsl[(r + mrow + 16) * LSTR + col] = acc1[r];
    }
  }
  __syncthreads();

  // ---- per-row masked log-softmax -> chosen log-prob (wave 0, one row/lane)
  if (tid < TM) {
    int m    = tid;
    int grow = row0 + m;
    int b = grow / Tn, t = grow % Tn;
    int Kb     = Karr[b];
    int active = (t <= Kb);
    int action = (t < Kb) ? pos[b * Nn + t] : Nn;          // flip col or finish(128)
    int lf     = (t == 0) ? -1 : pos[b * Nn + t - 1];      // last_flip before step t
    const float* lr = &ldsl[m * LSTR];
    float mx = -3.0e38f;
    for (int n = 0; n <= Nn; ++n) {
      bool valid = (n == Nn) || (n > lf);                  // finish col never masked
      if (valid) { float v = lr[n]; mx = v > mx ? v : mx; }
    }
    float s = 0.0f;
    for (int n = 0; n <= Nn; ++n) {
      bool valid = (n == Nn) || (n > lf);
      if (valid) s += __expf(lr[n] - mx);
    }
    float chosen = lr[action];
    lp_rows[grow] = active ? (chosen - mx - __logf(s)) : 0.0f;
  }
}

// ---------------------------------------------------------------------------
// Kernel 4: deterministic serial reduction of the 129 step contributions per b.
__global__ void reduce_lp(const float* __restrict__ lp_rows, float* __restrict__ out) {
  int b = threadIdx.x;  // 128 threads, 1 block
  float s = 0.0f;
  for (int t = 0; t < Tn; ++t) s += lp_rows[b * Tn + t];
  out[b] = s;
}

// ---------------------------------------------------------------------------
extern "C" void kernel_launch(void* const* d_in, const int* in_sizes, int n_in,
                              void* d_out, int out_size, void* d_ws, size_t ws_size,
                              hipStream_t stream) {
  (void)in_sizes; (void)n_in; (void)out_size; (void)ws_size;
  const float* one_hot = (const float*)d_in[0];
  const float* W1 = (const float*)d_in[1];
  const float* b1 = (const float*)d_in[2];
  const float* W2 = (const float*)d_in[3];
  const float* b2 = (const float*)d_in[4];
  const float* W3 = (const float*)d_in[5];
  const float* b3 = (const float*)d_in[6];

  char* ws = (char*)d_ws;
  unsigned short* W1b    = (unsigned short*)(ws + 0);        // 512*128*2  = 131072
  unsigned short* W2b    = (unsigned short*)(ws + 131072);   // 512*512*2  = 524288
  unsigned short* W3b    = (unsigned short*)(ws + 655360);   // 144*512*2  = 147456
  int*            samples= (int*)(ws + 802816);              // 128*128*4  = 65536
  int*            rank   = (int*)(ws + 868352);              // 128*128*4  = 65536
  int*            pos    = (int*)(ws + 933888);              // 128*128*4  = 65536
  int*            Karr   = (int*)(ws + 999424);              // 128*4      = 512
  float*          lp_rows= (float*)(ws + 999936);            // 16512*4    = 66048

  prep_weights<<<512, 256, 0, stream>>>(W1, W2, W3, W1b, W2b, W3b);
  prep_traj<<<Bn, Nn, 0, stream>>>(one_hot, samples, rank, pos, Karr);
  mlp_fused<<<ROWS / TM, 128, 0, stream>>>(W1b, W2b, W3b, b1, b2, b3,
                                           samples, rank, pos, Karr, lp_rows);
  reduce_lp<<<1, Bn, 0, stream>>>(lp_rows, (float*)d_out);
}
